// VideoMambaBlock_10230612099832
// MI455X (gfx1250) — matrix operand order repaired and checked by reading, not compiled, not run on hardware
//
#include <hip/hip_runtime.h>

// ---------------- model dims ----------------
#define IN_DIM   768
#define HID      1024
#define CLIP     512
#define NG       4
#define DS       64
#define DC       4
#define HD       64
#define DIN      2048      // 2*HID
#define NH       32        // DIN/HD
#define CONV_DIM 2560      // DIN + 2*NG*DS
#define DINP     4640      // 2*DIN + 2*NG*DS + NH
#define BATCH    1024
#define EPS      1e-5f

// ---------------- vector types ----------------
typedef __attribute__((ext_vector_type(16))) __bf16      bf16x16;
typedef __attribute__((ext_vector_type(8)))  float        f32x8;
typedef __attribute__((ext_vector_type(4)))  float        f32x4;
typedef __attribute__((ext_vector_type(4)))  unsigned int u32x4;

union ABfrag { u32x4 q[2]; bf16x16 v; };

// ---------------- helpers ----------------
__device__ __forceinline__ unsigned short f2bf(float f) {
    unsigned int u = __float_as_uint(f);
    u += 0x7fffu + ((u >> 16) & 1u);      // round-to-nearest-even
    return (unsigned short)(u >> 16);
}
__device__ __forceinline__ float sigmoidf_(float x) { return 1.0f / (1.0f + __expf(-x)); }
__device__ __forceinline__ float siluf_(float x)    { return x * sigmoidf_(x); }
__device__ __forceinline__ float softplusf_(float x){ return (x > 20.0f) ? x : __logf(1.0f + __expf(x)); }

__device__ __forceinline__ float blockReduceSum(float v, float* sm) {
    int t = threadIdx.x;
    sm[t] = v; __syncthreads();
    for (int o = blockDim.x >> 1; o > 0; o >>= 1) {
        if (t < o) sm[t] += sm[t + o];
        __syncthreads();
    }
    float r = sm[0]; __syncthreads();
    return r;
}

// ---------------- 1. LayerNorm -> bf16 x ----------------
__global__ void ln_kernel(const float* __restrict__ ff,
                          const float* __restrict__ w, const float* __restrict__ b,
                          unsigned short* __restrict__ x16) {
    __shared__ float sm[256];
    int bi = blockIdx.x, t = threadIdx.x;
    const float* row = ff + (size_t)bi * IN_DIM;
    float ls = 0.f, lq = 0.f;
    for (int i = t; i < IN_DIM; i += 256) { float x = row[i]; ls += x; lq += x * x; }
    float mu  = blockReduceSum(ls, sm) * (1.0f / IN_DIM);
    float ex2 = blockReduceSum(lq, sm) * (1.0f / IN_DIM);
    float rs  = rsqrtf(ex2 - mu * mu + EPS);
    for (int i = t; i < IN_DIM; i += 256) {
        float x = (row[i] - mu) * rs * w[i] + b[i];
        x16[(size_t)bi * IN_DIM + i] = f2bf(x);
    }
}

// ---------------- 2. f32 -> bf16 cast ----------------
__global__ void cvt_bf16(const float* __restrict__ in, unsigned short* __restrict__ out, int n) {
    int i = blockIdx.x * blockDim.x + threadIdx.x;
    if (i < n) out[i] = f2bf(in[i]);
}

// ---------------- 3. WMMA bf16 GEMM: C[M,N] = A[M,K] * B[N,K]^T ----------------
// Each wave owns a 16x64 strip (4 accumulators); one A-fragment feeds 4 WMMAs.
// ep: 0 -> acc+bias ; 1 -> (acc+bias)*sigmoid(other) ; 2 -> acc+other ; 3 -> other*sigmoid(acc+bias)
__global__ void wmma_gemm(const unsigned short* __restrict__ A,
                          const unsigned short* __restrict__ B,
                          const float* __restrict__ bias,
                          const float* __restrict__ other,
                          float* __restrict__ Cf,
                          unsigned short* __restrict__ Cbf,
                          int M, int N, int K, int ep) {
    const int lane  = threadIdx.x & 31;
    const int wave  = threadIdx.x >> 5;
    const int Ntiles = (N + 15) >> 4;
    const int tile0 = (blockIdx.x * 4 + wave) * 4;   // first of 4 n-tiles for this wave
    if (tile0 >= Ntiles) return;                     // wave-uniform exit
    const int mtile = blockIdx.y;

    const int rowBase = mtile * 16;
    const int m    = lane & 15;                      // A row / B col within tile
    const int half = lane >> 4;

    const unsigned short* arow = A + (size_t)(rowBase + m) * K;
    const unsigned short* brow[4];
#pragma unroll
    for (int s = 0; s < 4; ++s) {
        int ts = tile0 + s; if (ts > Ntiles - 1) ts = Ntiles - 1;   // clamp: loads stay in-bounds
        brow[s] = B + (size_t)(ts * 16 + m) * K;
    }

    f32x8 acc[4];
#pragma unroll
    for (int s = 0; s < 4; ++s) acc[s] = (f32x8)0.0f;

    for (int k = 0; k < K; k += 32) {
        const int ka = k + half * 16;                // contiguous k order (matches B)
        const int kb = k + half * 16;                // contiguous k order (matches A)
        ABfrag af;
        af.q[0] = *(const u32x4*)(arow + ka);
        af.q[1] = *(const u32x4*)(arow + ka + 8);
        __builtin_prefetch(arow + ka + 64, 0, 1);
#pragma unroll
        for (int s = 0; s < 4; ++s) {
            ABfrag bf;
            bf.q[0] = *(const u32x4*)(brow[s] + kb);
            bf.q[1] = *(const u32x4*)(brow[s] + kb + 8);
            acc[s] = __builtin_amdgcn_wmma_f32_16x16x32_bf16(false, af.v, false, bf.v,
                                                             (short)0, acc[s], false, false);
        }
    }

    const int n = lane & 15;
#pragma unroll
    for (int s = 0; s < 4; ++s) {
        if (tile0 + s >= Ntiles) break;              // wave-uniform
        const int col = (tile0 + s) * 16 + n;
        const float bv = bias ? bias[col] : 0.0f;
#pragma unroll
        for (int i = 0; i < 8; ++i) {
            const int row = rowBase + i + half * 8;  // C/D layout: VGPR i -> M=i / M=i+8
            const size_t idx = (size_t)row * N + col;
            float v = acc[s][i] + bv;
            if      (ep == 1) v = v * sigmoidf_(other[idx]);
            else if (ep == 2) v = v + other[idx];
            else if (ep == 3) v = other[idx] * sigmoidf_(v);
            if (Cf)  Cf[idx]  = v;
            if (Cbf) Cbf[idx] = f2bf(v);
        }
    }
}

// ---------------- 4. conv shift + depthwise conv + SiLU ----------------
__global__ void conv_kernel(const float* __restrict__ conv_state,
                            const float* __restrict__ zxbcdt,
                            const float* __restrict__ conv_w,
                            const float* __restrict__ conv_b,
                            float* __restrict__ out_conv,     // [B, CONV_DIM, DC]
                            float* __restrict__ xbc_act) {    // [B, CONV_DIM]
    int idx = blockIdx.x * blockDim.x + threadIdx.x;
    if (idx >= BATCH * CONV_DIM) return;
    int b = idx / CONV_DIM, c = idx % CONV_DIM;
    f32x4 cs = *(const f32x4*)(conv_state + (size_t)idx * DC);   // 16B aligned
    f32x4 w  = *(const f32x4*)(conv_w + (size_t)c * DC);
    float t3 = zxbcdt[(size_t)b * DINP + DIN + c];
    float accv = cs.y * w.x + cs.z * w.y + cs.w * w.z + t3 * w.w + conv_b[c];
    f32x4 nc; nc.x = cs.y; nc.y = cs.z; nc.z = cs.w; nc.w = t3;
    *(f32x4*)(out_conv + (size_t)idx * DC) = nc;
    xbc_act[idx] = siluf_(accv);
}

// ---------------- 5. SSM state update + contraction (bandwidth-critical) ----------------
// One workgroup per (head, batch). 256 threads; thread t owns 16 state elems as 4x float4.
__global__ void ssm_kernel(const float* __restrict__ ssm_state,
                           const float* __restrict__ xbc,       // [B, CONV_DIM]
                           const float* __restrict__ zxbcdt,    // dt at [.., DIN+CONV_DIM+h]
                           const float* __restrict__ A_log,
                           const float* __restrict__ Dp,
                           const float* __restrict__ dt_bias,
                           float* __restrict__ out_ssm,         // [B, NH, HD, DS]
                           float* __restrict__ yraw) {          // [B, DIN]
    __shared__ float sx[HD], sB[DS], sC[DS], ypart[HD * 4];
    const int h = blockIdx.x, b = blockIdx.y, t = threadIdx.x;
    const size_t xoff = (size_t)b * CONV_DIM;
    if (t < 64)        sx[t]      = xbc[xoff + h * HD + t];
    else if (t < 128)  sB[t - 64] = xbc[xoff + DIN + (h >> 3) * DS + (t - 64)];
    else if (t < 192)  sC[t - 128]= xbc[xoff + DIN + NG * DS + (h >> 3) * DS + (t - 128)];
    __syncthreads();

    float dt = softplusf_(zxbcdt[(size_t)b * DINP + DIN + CONV_DIM + h] + dt_bias[h]);
    float dA = __expf(dt * (-__expf(A_log[h])));

    const int p  = t >> 2;
    const int n0 = (t & 3) * 16;
    const size_t base = (((size_t)b * NH + h) * HD + p) * DS + n0;   // 64B aligned
    const f32x4* sp4 = (const f32x4*)(ssm_state + base);
    f32x4*       dp4 = (f32x4*)(out_ssm + base);
    const float xp = sx[p] * dt;
    float part = 0.f;
#pragma unroll
    for (int j = 0; j < 4; ++j) {
        const int n = n0 + j * 4;
        f32x4 sv = sp4[j];
        f32x4 s;
        s.x = sv.x * dA + xp * sB[n + 0];
        s.y = sv.y * dA + xp * sB[n + 1];
        s.z = sv.z * dA + xp * sB[n + 2];
        s.w = sv.w * dA + xp * sB[n + 3];
        dp4[j] = s;
        part += s.x * sC[n + 0] + s.y * sC[n + 1] + s.z * sC[n + 2] + s.w * sC[n + 3];
    }
    ypart[p * 4 + (t & 3)] = part;
    __syncthreads();
    if (t < 64) {
        float yv = ypart[t * 4] + ypart[t * 4 + 1] + ypart[t * 4 + 2] + ypart[t * 4 + 3]
                 + Dp[h] * sx[t];
        yraw[(size_t)b * DIN + h * HD + t] = yv;
    }
}

// ---------------- 6. gated group RMSNorm -> bf16 ----------------
__global__ void rmsnorm_kernel(const float* __restrict__ yraw,
                               const float* __restrict__ zxbcdt,   // z = [:, :DIN]
                               const float* __restrict__ norm_w,
                               unsigned short* __restrict__ y16) {
    __shared__ float sm[256];
    const int g = blockIdx.x, b = blockIdx.y, t = threadIdx.x;
    const int GW = DIN / NG;                    // 512
    const int c0 = g * GW + t, c1 = c0 + 256;
    float z0 = zxbcdt[(size_t)b * DINP + c0];
    float z1 = zxbcdt[(size_t)b * DINP + c1];
    float v0 = yraw[(size_t)b * DIN + c0] * siluf_(z0);
    float v1 = yraw[(size_t)b * DIN + c1] * siluf_(z1);
    float ssum = blockReduceSum(v0 * v0 + v1 * v1, sm);
    float rms = rsqrtf(ssum * (1.0f / GW) + EPS);
    y16[(size_t)b * DIN + c0] = f2bf(v0 * rms * norm_w[c0]);
    y16[(size_t)b * DIN + c1] = f2bf(v1 * rms * norm_w[c1]);
}

// ---------------- host: orchestrate ----------------
extern "C" void kernel_launch(void* const* d_in, const int* in_sizes, int n_in,
                              void* d_out, int out_size, void* d_ws, size_t ws_size,
                              hipStream_t stream) {
    (void)in_sizes; (void)n_in; (void)out_size; (void)ws_size;

    const float* frame_feat  = (const float*)d_in[0];
    const float* conv_state  = (const float*)d_in[1];
    const float* ssm_state   = (const float*)d_in[2];
    const float* ln_w        = (const float*)d_in[3];
    const float* ln_b        = (const float*)d_in[4];
    const float* w_in_gate   = (const float*)d_in[5];
    const float* b_in_gate   = (const float*)d_in[6];
    const float* w_input_proj= (const float*)d_in[7];
    const float* b_input_proj= (const float*)d_in[8];
    const float* w_in_proj   = (const float*)d_in[9];
    const float* conv_w      = (const float*)d_in[10];
    const float* conv_b      = (const float*)d_in[11];
    const float* A_log       = (const float*)d_in[12];
    const float* Dp          = (const float*)d_in[13];
    const float* dt_bias     = (const float*)d_in[14];
    const float* norm_w      = (const float*)d_in[15];
    const float* w_out_proj  = (const float*)d_in[16];
    const float* w_out_gate  = (const float*)d_in[17];
    const float* b_out_gate  = (const float*)d_in[18];
    const float* w_proj      = (const float*)d_in[19];
    const float* b_proj      = (const float*)d_in[20];

    float* out       = (float*)d_out;
    float* out_clip  = out;                                     // [B, CLIP]
    float* out_conv  = out + (size_t)BATCH * CLIP;              // [B, CONV_DIM, DC]
    float* out_ssm   = out_conv + (size_t)BATCH * CONV_DIM * DC;// [B, NH, HD, DS]

    // workspace bump allocator
    char* base = (char*)d_ws;
    size_t off = 0;
    auto alloc = [&](size_t bytes) -> char* {
        char* p = base + off;
        off = (off + bytes + 255) & ~(size_t)255;
        return p;
    };
    unsigned short* x16     = (unsigned short*)alloc((size_t)BATCH * IN_DIM * 2);
    unsigned short* wig16   = (unsigned short*)alloc((size_t)HID * IN_DIM * 2);
    unsigned short* wip16   = (unsigned short*)alloc((size_t)HID * IN_DIM * 2);
    unsigned short* winp16  = (unsigned short*)alloc((size_t)DINP * HID * 2);
    unsigned short* wop16   = (unsigned short*)alloc((size_t)HID * DIN * 2);
    unsigned short* wog16   = (unsigned short*)alloc((size_t)HID * HID * 2);
    unsigned short* wp16    = (unsigned short*)alloc((size_t)CLIP * HID * 2);
    float*          glogits = (float*)alloc((size_t)BATCH * HID * 4);
    float*          gated   = (float*)alloc((size_t)BATCH * HID * 4);
    unsigned short* gated16 = (unsigned short*)alloc((size_t)BATCH * HID * 2);
    float*          zxbcdt  = (float*)alloc((size_t)BATCH * DINP * 4);
    float*          xbc     = (float*)alloc((size_t)BATCH * CONV_DIM * 4);
    float*          yraw    = (float*)alloc((size_t)BATCH * DIN * 4);
    unsigned short* y16     = (unsigned short*)alloc((size_t)BATCH * DIN * 2);
    float*          outf    = (float*)alloc((size_t)BATCH * HID * 4);
    unsigned short* out16   = (unsigned short*)alloc((size_t)BATCH * HID * 2);
    unsigned short* out216  = (unsigned short*)alloc((size_t)BATCH * HID * 2);

    // 1) LayerNorm -> bf16 activations
    ln_kernel<<<BATCH, 256, 0, stream>>>(frame_feat, ln_w, ln_b, x16);

    // 2) weights -> bf16
    auto cvt = [&](const float* src, unsigned short* dst, size_t n) {
        cvt_bf16<<<(unsigned)((n + 255) / 256), 256, 0, stream>>>(src, dst, (int)n);
    };
    cvt(w_in_gate,    wig16,  (size_t)HID * IN_DIM);
    cvt(w_input_proj, wip16,  (size_t)HID * IN_DIM);
    cvt(w_in_proj,    winp16, (size_t)DINP * HID);
    cvt(w_out_proj,   wop16,  (size_t)HID * DIN);
    cvt(w_out_gate,   wog16,  (size_t)HID * HID);
    cvt(w_proj,       wp16,   (size_t)CLIP * HID);

    auto gemm = [&](const unsigned short* A, const unsigned short* B,
                    const float* bias, const float* other,
                    float* Cf, unsigned short* Cbf, int M, int N, int K, int ep) {
        int Ntiles = (N + 15) / 16;
        int wavesNeeded = (Ntiles + 3) / 4;          // each wave covers 4 n-tiles
        dim3 grid((wavesNeeded + 3) / 4, M / 16);    // 4 waves per block
        wmma_gemm<<<grid, 128, 0, stream>>>(A, B, bias, other, Cf, Cbf, M, N, K, ep);
    };

    // 3) gate logits; 4) gated = (x@Wip^T+b) * sigmoid(logits)
    gemm(x16, wig16, b_in_gate,    nullptr, glogits, nullptr, BATCH, HID, IN_DIM, 0);
    gemm(x16, wip16, b_input_proj, glogits, gated,   gated16, BATCH, HID, IN_DIM, 1);
    // 5) zxbcdt = gated @ w_in_proj^T
    gemm(gated16, winp16, nullptr, nullptr, zxbcdt, nullptr, BATCH, DINP, HID, 0);

    // 6) conv shift + activation (writes new_conv output)
    conv_kernel<<<(BATCH * CONV_DIM + 255) / 256, 256, 0, stream>>>(
        conv_state, zxbcdt, conv_w, conv_b, out_conv, xbc);

    // 7) SSM update + y (writes new_ssm output) — single pass over 1 GB of state
    ssm_kernel<<<dim3(NH, BATCH), 256, 0, stream>>>(
        ssm_state, xbc, zxbcdt, A_log, Dp, dt_bias, out_ssm, yraw);

    // 8) gated group RMSNorm -> bf16
    rmsnorm_kernel<<<dim3(NG, BATCH), 256, 0, stream>>>(yraw, zxbcdt, norm_w, y16);

    // 9) out = y@Wout^T + gated  (f32 + bf16)
    gemm(y16, wop16, nullptr, gated, outf, out16, BATCH, HID, DIN, 2);
    // 10) out2 = out * sigmoid(out@Wog^T + b)
    gemm(out16, wog16, b_out_gate, outf, nullptr, out216, BATCH, HID, HID, 3);
    // 11) clip_emb = out2 @ Wp^T + b  (writes clip output)
    gemm(out216, wp16, b_proj, nullptr, out_clip, nullptr, BATCH, CLIP, HID, 0);
}
